// ParallelVocoderNetwork_76381698392917
// MI455X (gfx1250) — compile-verified, matrix operand order checked by reading, not run on hardware
//
#include <hip/hip_runtime.h>
#include <hip/hip_fp16.h>

typedef __attribute__((ext_vector_type(16))) _Float16 v16h;
typedef __attribute__((ext_vector_type(8)))  float    v8f;

typedef __attribute__((address_space(1))) int gint_as1;
typedef __attribute__((address_space(3))) int lint_as3;

#define RF    1024
#define US    200
#define MGCN  60
#define FSN   64
#define NSN   4
#define NLN   10
#define NWIN  1600
#define SIGLEN 2624
#define SCALEC 0.70710678118f

// ---------------- workspace layout (in d_ws) ----------------
// cond  : 96000 f32   (1600 x 60)
// prevA : 2624  f32
// prevB : 2624  f32
// wpk   : 442368 u32  (4 stages x 9 layers x 48 tiles x 32 lanes x 8 dwords)
#define WS_COND   0
#define WS_PREVA  96000
#define WS_PREVB  (96000 + 2624)
#define WS_WPK    (96000 + 2624 + 2624)
#define WPK_DWORDS (4*9*48*32*8)

#if __has_builtin(__builtin_amdgcn_global_load_async_to_lds_b32)
#define HAVE_ASYNC_LDS 1
#endif

__device__ __forceinline__ float sigm(float x) { return 1.0f / (1.0f + expf(-x)); }

__device__ __forceinline__ void wait_asynccnt0() {
#if defined(HAVE_ASYNC_LDS)
#if __has_builtin(__builtin_amdgcn_s_wait_asynccnt)
  __builtin_amdgcn_s_wait_asynccnt(0);
#else
  asm volatile("s_wait_asynccnt 0x0" ::: "memory");
#endif
#endif
}

// ---------------------------------------------------------------------------
// Prep 1: cond = tanh(mgc @ cond_w.T + cond_b).reshape(1600, 60)
// ---------------------------------------------------------------------------
__global__ __launch_bounds__(256) void cond_prep_kernel(
    const float* __restrict__ mgc, const float* __restrict__ cond_w,
    const float* __restrict__ cond_b, float* __restrict__ cond)
{
  int idx = blockIdx.x * 256 + threadIdx.x;
  if (idx >= NWIN * MGCN) return;
  int n = idx / MGCN, c = idx % MGCN;
  int f = n / US, u = n % US;
  int row = u * MGCN + c;
  float v = cond_b[row];
  const float* w = &cond_w[row * MGCN];
  const float* m = &mgc[f * MGCN];
  for (int k = 0; k < MGCN; ++k) v += m[k] * w[k];
  cond[idx] = tanhf(v);
}

// ---------------------------------------------------------------------------
// Prep 2: pack convs_w (f32) into f16 WMMA-B lane layout.
// Tile (kt,nt): 32 lanes x 8 dwords. lane: n = nt*16+(lane&15), hi = lane>>4.
// dword d holds halves h=2d,2d+1 with k = (h&7) + ((h>>3)<<4) + (hi<<3),
// global c = kt*16 + (k>>1), tap t = k&1, B[k][n] = w[g= n>>6][o=n&63][c][t].
// ---------------------------------------------------------------------------
__global__ __launch_bounds__(256) void wpack_kernel(
    const float* __restrict__ convs_w, unsigned int* __restrict__ wpk)
{
  int idx = blockIdx.x * 256 + threadIdx.x;
  if (idx >= WPK_DWORDS) return;
  int d     = idx & 7;
  int lanei = (idx >> 3) & 31;
  int tile  = (idx >> 8) % 48;
  int sl    = (idx >> 8) / 48;
  int kt = tile / 12, nt = tile % 12;
  int s = sl / 9, lm1 = sl % 9;
  int nn = nt * 16 + (lanei & 15);
  int hh = lanei >> 4;
  int g = nn >> 6, o = nn & 63;
  union { _Float16 h[2]; unsigned int u; } P;
  #pragma unroll
  for (int half = 0; half < 2; ++half) {
    int hb = 2 * d + half;
    int k  = (hb & 7) + ((hb >> 3) << 4) + (hh << 3);
    int c  = kt * 16 + (k >> 1);
    int t  = k & 1;
    float v = convs_w[((((s * 9 + lm1) * 3 + g) * 64 + o) * 64 + c) * 2 + t];
    P.h[half] = (_Float16)v;
  }
  wpk[idx] = P.u;
}

// ---------------------------------------------------------------------------
// Prep 3: init ping-pong signal buffers.
// ---------------------------------------------------------------------------
__global__ __launch_bounds__(256) void init_kernel(
    const float* __restrict__ noise, float* __restrict__ prevA, float* __restrict__ prevB)
{
  int i = blockIdx.x * 256 + threadIdx.x;
  if (i < SIGLEN) prevA[i] = noise[i];
  if (i < RF)     prevB[i] = noise[i];
}

// ---------------------------------------------------------------------------
// Stage kernel: one block per window n. Full 10-layer gated conv tree in LDS
// (f16 activations, WMMA f32<-f16 for layers 1..9), then the 64->256->1 head.
// ---------------------------------------------------------------------------
#define SM_HA    0          // 64*512 f16 = 65536 B
#define SM_HB    65536      // 64*256 f16 = 32768 B
#define SM_X     98304      // 1024 f32  = 4096 B
#define SM_COND  102400     // 60 f32 (pad 256)
#define SM_ICGC  102656     // 192 f32 = 768 B
#define SM_PRE   103424     // 256 f32 = 1024 B
#define SM_RED   104448     // 128 f32 = 512 B
#define SM_BYTES 104960

__global__ __launch_bounds__(256) void stage_kernel(
    int s,
    const float* __restrict__ prevIn,
    float* __restrict__ prevOut,                 // unused when s==3
    const float* __restrict__ cond,
    const unsigned int* __restrict__ wpk,
    const float* __restrict__ conv0_w,
    const float* __restrict__ conv_b,
    const float* __restrict__ ccw,
    const float* __restrict__ ccb,
    const float* __restrict__ pre_w,
    const float* __restrict__ pre_b,
    const float* __restrict__ mean_w,
    const float* __restrict__ mean_b,
    const float* __restrict__ std_w,
    const float* __restrict__ std_b,
    float* __restrict__ out)
{
  extern __shared__ char smem[];
  _Float16* hAbuf = (_Float16*)(smem + SM_HA);
  _Float16* hBbuf = (_Float16*)(smem + SM_HB);
  float* sx    = (float*)(smem + SM_X);
  float* scond = (float*)(smem + SM_COND);
  float* sicgc = (float*)(smem + SM_ICGC);
  float* spre  = (float*)(smem + SM_PRE);
  float* sred  = (float*)(smem + SM_RED);

  const int tid = threadIdx.x;
  const int n   = blockIdx.x;
  const int lane = tid & 31;
  const int wv   = tid >> 5;
  const int m16  = lane & 15;
  const int hi   = lane >> 4;

  if (tid < MGCN) scond[tid] = cond[n * MGCN + tid];

  // Window load: async global->LDS (ASYNCcnt path) when available.
#if defined(HAVE_ASYNC_LDS)
  #pragma unroll
  for (int i = 0; i < RF / 256; ++i) {
    __builtin_amdgcn_global_load_async_to_lds_b32(
        (gint_as1*)(prevIn + n + i * 256 + tid),
        (lint_as3*)(sx + i * 256 + tid),
        0, 0);
  }
  wait_asynccnt0();
#else
  for (int i = tid; i < RF; i += 256) sx[i] = prevIn[n + i];
#endif
  __syncthreads();

  // ---------------- layer 0 (C=1 -> 64, L 1024 -> 512), VALU ----------------
  if (tid < 192) {
    int g = tid >> 6, o = tid & 63;
    float v = conv_b[((s * NLN + 0) * 3 + g) * 64 + o];
    if (g < 2) {
      v += ccb[((s * NLN + 0) * 2 + g) * 64 + o];
      const float* w = &ccw[(((s * NLN + 0) * 2 + g) * 64 + o) * MGCN];
      for (int c = 0; c < MGCN; ++c) v += scond[c] * w[c];
    }
    sicgc[tid] = v;
  }
  __syncthreads();
  {
    int o  = tid & 63;
    int j0 = tid >> 6;
    float a0 = conv0_w[((s * 3 + 0) * 64 + o) * 2 + 0];
    float a1 = conv0_w[((s * 3 + 0) * 64 + o) * 2 + 1];
    float b0 = conv0_w[((s * 3 + 1) * 64 + o) * 2 + 0];
    float b1 = conv0_w[((s * 3 + 1) * 64 + o) * 2 + 1];
    float c0 = conv0_w[((s * 3 + 2) * 64 + o) * 2 + 0];
    float c1 = conv0_w[((s * 3 + 2) * 64 + o) * 2 + 1];
    float ic = sicgc[o], gc = sicgc[64 + o], rb = sicgc[128 + o];
    for (int j = j0; j < 512; j += 4) {
      float xa = sx[2 * j], xb = sx[2 * j + 1];
      float r = (tanhf(a0 * xa + a1 * xb + ic) * sigm(b0 * xa + b1 * xb + gc)
                 + (c0 * xa + c1 * xb + rb)) * SCALEC;
      hAbuf[o * 512 + j] = (_Float16)r;
    }
  }

  // ---------------- layers 1..9: WMMA M x 192 x 128 ----------------
  for (int l = 1; l < NLN; ++l) {
    __syncthreads();
    // per-layer cond projection + conv biases into sicgc[0..191]
    if (tid < 192) {
      int g = tid >> 6, o = tid & 63;
      float v = conv_b[((s * NLN + l) * 3 + g) * 64 + o];
      if (g < 2) {
        v += ccb[((s * NLN + l) * 2 + g) * 64 + o];
        const float* w = &ccw[(((s * NLN + l) * 2 + g) * 64 + o) * MGCN];
        for (int c = 0; c < MGCN; ++c) v += scond[c] * w[c];
      }
      sicgc[tid] = v;
    }
    const int M = 256 >> (l - 1);                 // output positions this layer
    const _Float16* hin = (l & 1) ? hAbuf : hBbuf;
    _Float16* hout      = (l & 1) ? hBbuf : hAbuf;
    const unsigned int* hin_u = (const unsigned int*)hin;
    const unsigned int* wbase = wpk + (unsigned)((s * 9 + (l - 1)) * 48) * 256u;
    if (l < NLN - 1) __builtin_prefetch(wbase + 48 * 256 + tid * 16, 0, 0);
    __syncthreads();

    const int Mtiles = (M + 15) >> 4;
    const int ngroups = Mtiles * 4;
    for (int grp = wv; grp < ngroups; grp += 8) {
      const int mt = grp >> 2, ntg = grp & 3;
      const int jrow = mt * 16 + m16;             // A row = output position
      v8f acc0 = {}; v8f acc1 = {}; v8f acc2 = {};
      #pragma unroll
      for (int kt = 0; kt < 4; ++kt) {
        union { unsigned int u[8]; v16h h; } A;
        #pragma unroll
        for (int d = 0; d < 8; ++d) {
          int c = kt * 16 + (d & 3) + ((d >> 2) << 3) + (hi << 2);
          A.u[d] = hin_u[c * M + jrow];           // two taps (2j,2j+1) in one dword
        }
        const uint4* bp0 = (const uint4*)(wbase + (unsigned)(kt * 12 + ntg + 0) * 256u + lane * 8);
        const uint4* bp1 = (const uint4*)(wbase + (unsigned)(kt * 12 + ntg + 4) * 256u + lane * 8);
        const uint4* bp2 = (const uint4*)(wbase + (unsigned)(kt * 12 + ntg + 8) * 256u + lane * 8);
        union { uint4 q[2]; v16h h; } B;
        B.q[0] = bp0[0]; B.q[1] = bp0[1];
        acc0 = __builtin_amdgcn_wmma_f32_16x16x32_f16(false, A.h, false, B.h, (short)0, acc0, false, false);
        B.q[0] = bp1[0]; B.q[1] = bp1[1];
        acc1 = __builtin_amdgcn_wmma_f32_16x16x32_f16(false, A.h, false, B.h, (short)0, acc1, false, false);
        B.q[0] = bp2[0]; B.q[1] = bp2[1];
        acc2 = __builtin_amdgcn_wmma_f32_16x16x32_f16(false, A.h, false, B.h, (short)0, acc2, false, false);
      }
      // gated combine: out = (tanh(inp+ic)*sigmoid(gate+gc) + res+rb)*SCALE
      const int och = ntg * 16 + m16;             // D column = output channel
      const float ic = sicgc[och], gc = sicgc[64 + och], rb = sicgc[128 + och];
      if (M >= 16) {
        unsigned int pk[4];
        #pragma unroll
        for (int v = 0; v < 8; v += 2) {
          union { _Float16 h[2]; unsigned int u; } P;
          float r0 = (tanhf(acc0[v] + ic) * sigm(acc1[v] + gc) + acc2[v] + rb) * SCALEC;
          float r1 = (tanhf(acc0[v + 1] + ic) * sigm(acc1[v + 1] + gc) + acc2[v + 1] + rb) * SCALEC;
          P.h[0] = (_Float16)r0; P.h[1] = (_Float16)r1;
          pk[v >> 1] = P.u;
        }
        uint4 q; q.x = pk[0]; q.y = pk[1]; q.z = pk[2]; q.w = pk[3];
        *(uint4*)(hout + och * M + mt * 16 + hi * 8) = q;   // 8 contiguous f16
      } else {
        #pragma unroll
        for (int v = 0; v < 8; ++v) {
          int jj = hi * 8 + v;
          if (jj < M) {
            float r = (tanhf(acc0[v] + ic) * sigm(acc1[v] + gc) + acc2[v] + rb) * SCALEC;
            hout[och * M + jj] = (_Float16)r;
          }
        }
      }
    }
  }
  __syncthreads();

  // ---------------- head: h10 (in hBbuf[0..63]) -> pre(256) -> mean/logvar ----
  {
    float a = pre_b[s * 256 + tid];
    const float* w = &pre_w[(s * 256 + tid) * 64];
    for (int c = 0; c < 64; ++c) a += (float)hBbuf[c] * w[c];
    spre[tid] = a > 0.0f ? a : 0.0f;
  }
  __syncthreads();
  if (tid < 128) {
    int which = tid >> 6, t = tid & 63;
    const float* w = which ? &std_w[s * 256] : &mean_w[s * 256];
    float a = 0.0f;
    for (int i = t; i < 256; i += 64) a += spre[i] * w[i];
    sred[tid] = a;
  }
  __syncthreads();
  if (tid == 0) {
    float mean = mean_b[s], lv = std_b[s];
    for (int i = 0; i < 64; ++i) { mean += sred[i]; lv += sred[64 + i]; }
    float eps = prevIn[RF + n];
    float nv  = eps * expf(0.5f * lv) + mean;
    if (s < NSN - 1) prevOut[RF + n] = nv;
    if (n >= NWIN - RF) out[3 * NWIN + s * RF + (n - (NWIN - RF))] = nv;  // tails
    if (s == NSN - 1) {
      out[n] = nv;                 // final prev_x
      out[NWIN + n] = mean;        // mean
      out[2 * NWIN + n] = lv;      // logvar
    }
  }
}

// ---------------------------------------------------------------------------
extern "C" void kernel_launch(void* const* d_in, const int* in_sizes, int n_in,
                              void* d_out, int out_size, void* d_ws, size_t ws_size,
                              hipStream_t stream) {
  (void)in_sizes; (void)n_in; (void)out_size; (void)ws_size;
  const float* mgc     = (const float*)d_in[0];
  const float* noise   = (const float*)d_in[1];
  const float* cond_w  = (const float*)d_in[2];
  const float* cond_b  = (const float*)d_in[3];
  const float* conv0_w = (const float*)d_in[4];
  const float* convs_w = (const float*)d_in[5];
  const float* conv_b  = (const float*)d_in[6];
  const float* ccw     = (const float*)d_in[7];
  const float* ccb     = (const float*)d_in[8];
  const float* pre_w   = (const float*)d_in[9];
  const float* pre_b   = (const float*)d_in[10];
  const float* mean_w  = (const float*)d_in[11];
  const float* mean_b  = (const float*)d_in[12];
  const float* std_w   = (const float*)d_in[13];
  const float* std_b   = (const float*)d_in[14];
  float* out = (float*)d_out;

  float* ws    = (float*)d_ws;
  float* cond  = ws + WS_COND;
  float* prevA = ws + WS_PREVA;
  float* prevB = ws + WS_PREVB;
  unsigned int* wpk = (unsigned int*)(ws + WS_WPK);

  cond_prep_kernel<<<(NWIN * MGCN + 255) / 256, 256, 0, stream>>>(mgc, cond_w, cond_b, cond);
  wpack_kernel<<<WPK_DWORDS / 256, 256, 0, stream>>>(convs_w, wpk);
  init_kernel<<<(SIGLEN + 255) / 256, 256, 0, stream>>>(noise, prevA, prevB);

  for (int s = 0; s < NSN; ++s) {
    const float* pin = (s & 1) ? prevB : prevA;
    float*       pot = (s & 1) ? prevA : prevB;
    stage_kernel<<<NWIN, 256, SM_BYTES, stream>>>(
        s, pin, pot, cond, wpk, conv0_w, conv_b, ccw, ccb,
        pre_w, pre_b, mean_w, mean_b, std_w, std_b, out);
  }
}
